// AutoGraphConstructionModule_55473797595943
// MI455X (gfx1250) — matrix operand order repaired and checked by reading, not compiled, not run on hardware
//
#include <hip/hip_runtime.h>
#include <hip/hip_bf16.h>
#include <math.h>

#define B_ 8
#define S_ 4096
#define D_ 1024
#define H_ 256
#define T_ 8
#define K_ENT 20
#define EPS_ 1e-5f
#define THRESH_ 0.4f
#define NSP 2048

typedef __bf16 v16bf __attribute__((ext_vector_type(16)));
typedef float  v8f   __attribute__((ext_vector_type(8)));

__device__ __forceinline__ __bf16 f2bf(float x) { return (__bf16)x; }
__device__ __forceinline__ float gelu_exact(float x) {
    return 0.5f * x * (1.0f + erff(x * 0.70710678118654752f));
}

// ---------------------------------------------------------------------------
// Kernel 0: one-time convert+tile of w1 (fp32 [1024,256]) into bf16
// tiles w1t[D/32][256][32]  (k-chunk outer, col, k-inner) so the scorer can
// stage B tiles with coalesced 128-bit async copies, zero conversions.
// ---------------------------------------------------------------------------
__global__ __launch_bounds__(256) void cvt_w1_kernel(
    const float* __restrict__ w1, __bf16* __restrict__ w1t)
{
    int idx = blockIdx.x * 256 + threadIdx.x;      // 0 .. D*H-1
    int kglob = idx >> 8;                          // 0..1023 (row of w1)
    int col   = idx & 255;
    int kc = kglob >> 5, kk = kglob & 31;
    w1t[(size_t)kc * (H_ * 32) + col * 32 + kk] = f2bf(w1[(size_t)kglob * H_ + col]);
}

// ---------------------------------------------------------------------------
// Kernel 1: fused scorer.  X[32768,1024] @ w1[1024,256] (bf16 WMMA) ->
// +b1 -> LayerNorm -> GELU -> @w2[256,1] + b2 -> sigmoid -> *mask -> scores.
// Block = 256 threads (8 waves), tile 64 rows x 256 cols, K-step 32.
// Tiles move global->LDS via GLOBAL_LOAD_ASYNC_TO_LDS_B128 (ASYNCcnt),
// double-buffered so the DMA overlaps the WMMAs; no VGPR staging at all.
// ---------------------------------------------------------------------------
__global__ __launch_bounds__(256) void scorer_kernel(
    const float* __restrict__ X, const int* __restrict__ mask,
    const __bf16* __restrict__ w1t, const float* __restrict__ b1,
    const float* __restrict__ g,  const float* __restrict__ be,
    const float* __restrict__ w2, const float* __restrict__ b2,
    float* __restrict__ scores)
{
    __shared__ __align__(16) float  Asf[2][64][36];   // fp32 A tiles (144B rows)
    __shared__ __align__(16) __bf16 Bst[2][256][40];  // bf16 B tiles (80B rows)
    __shared__ float b1s[H_], gs[H_], bes[H_], w2s[H_];
    __shared__ float redS[2][64], redQ[2][64], redC[2][64];

    const int tid  = threadIdx.x;
    const int wave = tid >> 5;
    const int lane = tid & 31;
    const int rg   = wave >> 1;        // row group 0..3 (16 rows each)
    const int cg   = wave & 1;         // col group 0..1 (128 cols each)
    const int half = lane >> 4;        // 0/1
    const int l15  = lane & 15;
    const int rowBase = blockIdx.x * 64;

    // A staging assignment: 4 threads/row, 8 consecutive K floats each (32B)
    const int rA  = tid >> 2;
    const int kcA = (tid & 3) * 8;

    for (int i = tid; i < H_; i += 256) {
        b1s[i] = b1[i]; gs[i] = g[i]; bes[i] = be[i]; w2s[i] = w2[i];
    }

    v8f c[8];
    #pragma unroll
    for (int t = 0; t < 8; ++t)
        #pragma unroll
        for (int e = 0; e < 8; ++e) c[t][e] = 0.0f;

    const float*  Arow = X + (size_t)(rowBase + rA) * D_ + kcA;
    const __bf16* Brow = w1t + tid * 32;              // 64B per thread per tile

    const uint32_t ldsA0 = (uint32_t)(uintptr_t)&Asf[0][rA][kcA];
    const uint32_t ldsA1 = (uint32_t)(uintptr_t)&Asf[1][rA][kcA];
    const uint32_t ldsB0 = (uint32_t)(uintptr_t)&Bst[0][tid][0];
    const uint32_t ldsB1 = (uint32_t)(uintptr_t)&Bst[1][tid][0];

    auto issueTile = [&](int i, int buf) {
        const char* gA = (const char*)(Arow + i * 32);
        const char* gB = (const char*)(Brow + (size_t)i * (H_ * 32));
        uint32_t lA = buf ? ldsA1 : ldsA0;
        uint32_t lB = buf ? ldsB1 : ldsB0;
        // A tile: 32 bytes (8 fp32) per thread
        asm volatile("global_load_async_to_lds_b128 %0, %1, off"
                     :: "v"(lA), "v"(gA) : "memory");
        asm volatile("global_load_async_to_lds_b128 %0, %1, off offset:16"
                     :: "v"(lA), "v"(gA) : "memory");
        // B tile: 64 bytes (32 bf16) per thread
        asm volatile("global_load_async_to_lds_b128 %0, %1, off"
                     :: "v"(lB), "v"(gB) : "memory");
        asm volatile("global_load_async_to_lds_b128 %0, %1, off offset:16"
                     :: "v"(lB), "v"(gB) : "memory");
        asm volatile("global_load_async_to_lds_b128 %0, %1, off offset:32"
                     :: "v"(lB), "v"(gB) : "memory");
        asm volatile("global_load_async_to_lds_b128 %0, %1, off offset:48"
                     :: "v"(lB), "v"(gB) : "memory");
    };

    issueTile(0, 0);
    asm volatile("s_wait_asynccnt 0x0" ::: "memory");
    __syncthreads();

    const int NT = D_ / 32;   // 32 K-tiles
    for (int i = 0; i < NT; ++i) {
        const int cur = i & 1;
        if (i + 1 < NT) issueTile(i + 1, cur ^ 1);   // DMA overlaps WMMAs below

        // A fragment: two contiguous 8-K fp32 runs -> 4x ds_load_b128 + cvt
        v16bf a;
        {
            const float* ap = &Asf[cur][rg * 16 + l15][half * 8];
            float4 x0 = *(const float4*)(ap);
            float4 x1 = *(const float4*)(ap + 4);
            float4 x2 = *(const float4*)(ap + 16);
            float4 x3 = *(const float4*)(ap + 20);
            a[0]  = f2bf(x0.x); a[1]  = f2bf(x0.y); a[2]  = f2bf(x0.z); a[3]  = f2bf(x0.w);
            a[4]  = f2bf(x1.x); a[5]  = f2bf(x1.y); a[6]  = f2bf(x1.z); a[7]  = f2bf(x1.w);
            a[8]  = f2bf(x2.x); a[9]  = f2bf(x2.y); a[10] = f2bf(x2.z); a[11] = f2bf(x2.w);
            a[12] = f2bf(x3.x); a[13] = f2bf(x3.y); a[14] = f2bf(x3.z); a[15] = f2bf(x3.w);
        }
        #pragma unroll
        for (int t = 0; t < 8; ++t) {
            const int col = cg * 128 + t * 16 + l15;
            v16bf bq;
            uint4* bp = (uint4*)&bq;
            bp[0] = *(const uint4*)&Bst[cur][col][half * 8];
            bp[1] = *(const uint4*)&Bst[cur][col][16 + half * 8];
            c[t] = __builtin_amdgcn_wmma_f32_16x16x32_bf16(
                false, a, false, bq, (short)0, c[t], false, false);
        }

        if (i + 1 < NT) {
            asm volatile("s_wait_asynccnt 0x0" ::: "memory");
            __syncthreads();
        }
    }

    // ---- epilogue: +b1, LayerNorm stats via shfl_xor + cross-wave LDS ----
    float p[8], q[8];
    #pragma unroll
    for (int v = 0; v < 8; ++v) { p[v] = 0.f; q[v] = 0.f; }
    #pragma unroll
    for (int t = 0; t < 8; ++t) {
        int j = cg * 128 + t * 16 + l15;
        float bb = b1s[j];
        #pragma unroll
        for (int v = 0; v < 8; ++v) {
            float x = c[t][v] + bb;
            c[t][v] = x;
            p[v] += x; q[v] += x * x;
        }
    }
    #pragma unroll
    for (int m = 1; m < 16; m <<= 1) {
        #pragma unroll
        for (int v = 0; v < 8; ++v) {
            p[v] += __shfl_xor(p[v], m, 32);
            q[v] += __shfl_xor(q[v], m, 32);
        }
    }
    if (l15 == 0) {
        #pragma unroll
        for (int v = 0; v < 8; ++v) {
            int row = rg * 16 + half * 8 + v;
            redS[cg][row] = p[v]; redQ[cg][row] = q[v];
        }
    }
    __syncthreads();
    float mu[8], rs[8];
    #pragma unroll
    for (int v = 0; v < 8; ++v) {
        int row = rg * 16 + half * 8 + v;
        float sm = redS[0][row] + redS[1][row];
        float qq = redQ[0][row] + redQ[1][row];
        float m_ = sm * (1.0f / H_);
        float var = qq * (1.0f / H_) - m_ * m_;
        mu[v] = m_; rs[v] = rsqrtf(var + EPS_);
    }
    float sc[8];
    #pragma unroll
    for (int v = 0; v < 8; ++v) sc[v] = 0.f;
    #pragma unroll
    for (int t = 0; t < 8; ++t) {
        int j = cg * 128 + t * 16 + l15;
        float gg = gs[j], bb = bes[j], ww = w2s[j];
        #pragma unroll
        for (int v = 0; v < 8; ++v) {
            float x = (c[t][v] - mu[v]) * rs[v] * gg + bb;
            sc[v] += gelu_exact(x) * ww;
        }
    }
    #pragma unroll
    for (int m = 1; m < 16; m <<= 1)
        #pragma unroll
        for (int v = 0; v < 8; ++v) sc[v] += __shfl_xor(sc[v], m, 32);
    if (l15 == 0) {
        #pragma unroll
        for (int v = 0; v < 8; ++v) redC[cg][rg * 16 + half * 8 + v] = sc[v];
    }
    __syncthreads();
    if (tid < 64) {
        int m = rowBase + tid;
        float s = redC[0][tid] + redC[1][tid] + b2[0];
        float sg = 1.0f / (1.0f + expf(-s));
        scores[m] = sg * (float)mask[m];
    }
}

// ---------------------------------------------------------------------------
// Kernel 2: per-batch span RLE + per-span stats + top-K selection.
// ---------------------------------------------------------------------------
__global__ __launch_bounds__(256) void span_kernel(
    const float* __restrict__ scores, const int* __restrict__ mask,
    float* __restrict__ escore_out, float* __restrict__ validk_out,
    int* __restrict__ metaStart, int* __restrict__ metaLen,
    float* __restrict__ metaValid)
{
    __shared__ int   spanLen[NSP];
    __shared__ float spanSum[NSP];
    __shared__ int   spanStart[NSP];
    __shared__ float selA[NSP];
    __shared__ int   scanB[257];
    __shared__ float bestV[256];
    __shared__ int   bestI[256];
    __shared__ int   topIdx[K_ENT];
    __shared__ float topVal[K_ENT];

    const int b = blockIdx.x, tid = threadIdx.x;
    for (int i = tid; i < NSP; i += 256) {
        spanLen[i] = 0; spanSum[i] = 0.f; spanStart[i] = 0; selA[i] = -1.f;
    }
    __syncthreads();

    const int TPT = S_ / 256;   // 16 tokens/thread
    const int s0 = tid * TPT;
    bool prevAbove = false;
    if (s0 > 0) {
        float ps = scores[b * S_ + s0 - 1];
        prevAbove = (ps >= THRESH_) && (mask[b * S_ + s0 - 1] > 0);
    }
    float sv[TPT]; bool ab[TPT];
    int cnt = 0; bool pa = prevAbove;
    #pragma unroll
    for (int i = 0; i < TPT; ++i) {
        float v = scores[b * S_ + s0 + i];
        bool a = (v >= THRESH_) && (mask[b * S_ + s0 + i] > 0);
        sv[i] = v; ab[i] = a;
        if (a && !pa) ++cnt;
        pa = a;
    }
    scanB[tid] = cnt;
    __syncthreads();
    if (tid == 0) {
        int run = 0;
        for (int i = 0; i < 256; ++i) { int c = scanB[i]; scanB[i] = run; run += c; }
        scanB[256] = run;
    }
    __syncthreads();
    int run = scanB[tid]; pa = prevAbove;
    #pragma unroll
    for (int i = 0; i < TPT; ++i) {
        if (ab[i] && !pa) { ++run; spanStart[run - 1] = s0 + i; }
        if (ab[i]) {
            atomicAdd(&spanLen[run - 1], 1);
            atomicAdd(&spanSum[run - 1], sv[i]);
        }
        pa = ab[i];
    }
    __syncthreads();
    for (int j = tid; j < NSP; j += 256) {
        int L = spanLen[j];
        float avg = spanSum[j] / fmaxf((float)L, 1.f);
        bool valid = (L > 0) && (avg >= THRESH_);
        selA[j] = valid ? avg : -1.f;
    }
    __syncthreads();
    for (int k = 0; k < K_ENT; ++k) {
        float bv = -3.f; int bi = 0;
        for (int j = tid * (NSP / 256); j < (tid + 1) * (NSP / 256); ++j) {
            float v = selA[j];
            if (v > bv) { bv = v; bi = j; }
        }
        bestV[tid] = bv; bestI[tid] = bi;
        __syncthreads();
        if (tid == 0) {
            float gv = -3.f; int gi = 0;
            for (int i = 0; i < 256; ++i)
                if (bestV[i] > gv) { gv = bestV[i]; gi = bestI[i]; }
            topIdx[k] = gi; topVal[k] = gv; selA[gi] = -2.f;
        }
        __syncthreads();
    }
    if (tid < K_ENT) {
        int j = topIdx[tid]; float v = topVal[tid];
        float vk = (v >= THRESH_) ? 1.f : 0.f;
        escore_out[b * K_ENT + tid] = v * vk;
        validk_out[b * K_ENT + tid] = vk;
        metaStart[b * K_ENT + tid] = spanStart[j];
        metaLen[b * K_ENT + tid] = spanLen[j];
        metaValid[b * K_ENT + tid] = vk;
    }
}

// ---------------------------------------------------------------------------
// Kernel 3: mean-pool hidden states over each selected span.
// ---------------------------------------------------------------------------
__global__ __launch_bounds__(256) void pool_kernel(
    const float* __restrict__ X, const int* __restrict__ metaStart,
    const int* __restrict__ metaLen, const float* __restrict__ metaValid,
    float* __restrict__ reps)
{
    const int e = blockIdx.x;
    const int b = e / K_ENT;
    const int start = metaStart[e];
    const int len = metaLen[e];
    const float scale = metaValid[e] / fmaxf((float)len, 1.f);
    for (int d = threadIdx.x; d < D_; d += 256) {
        float s = 0.f;
        for (int i = 0; i < len; ++i)
            s += X[((size_t)b * S_ + start + i) * D_ + d];
        reps[(size_t)e * D_ + d] = s * scale;
    }
}

// ---------------------------------------------------------------------------
// Kernel 4: enhancer MLP + type MLP + softmax/argmax per selected entity.
// ---------------------------------------------------------------------------
__global__ __launch_bounds__(256) void mlp_kernel(
    const float* __restrict__ reps, const float* __restrict__ metaValid,
    const float* __restrict__ ew1, const float* __restrict__ eb1,
    const float* __restrict__ eg,  const float* __restrict__ ebe,
    const float* __restrict__ ew2, const float* __restrict__ eb2,
    const float* __restrict__ tw1, const float* __restrict__ tb1,
    const float* __restrict__ tg,  const float* __restrict__ tbe,
    const float* __restrict__ tw2, const float* __restrict__ tb2,
    float* __restrict__ out_enh, float* __restrict__ out_logits,
    float* __restrict__ out_probs, float* __restrict__ out_types)
{
    __shared__ float rep[D_];
    __shared__ float hact[H_];
    __shared__ float ebuf[D_];
    __shared__ float red[16];
    __shared__ float stats[2];
    __shared__ float lg[T_];

    const int e = blockIdx.x, tid = threadIdx.x;
    const float valid = metaValid[e];
    for (int d = tid; d < D_; d += 256) rep[d] = reps[(size_t)e * D_ + d];
    __syncthreads();

    // enhancer layer 1: D->H (thread = one of 256 hidden units)
    float h = eb1[tid];
    for (int d = 0; d < D_; ++d) h += rep[d] * ew1[(size_t)d * H_ + tid];
    {   // LayerNorm over 256
        float s = h, q = h * h;
        for (int m = 1; m < 32; m <<= 1) { s += __shfl_xor(s, m, 32); q += __shfl_xor(q, m, 32); }
        if ((tid & 31) == 0) { red[tid >> 5] = s; red[8 + (tid >> 5)] = q; }
        __syncthreads();
        if (tid == 0) {
            float ss = 0.f, qq = 0.f;
            for (int i = 0; i < 8; ++i) { ss += red[i]; qq += red[8 + i]; }
            float mu = ss / H_;
            stats[0] = mu; stats[1] = rsqrtf(qq / H_ - mu * mu + EPS_);
        }
        __syncthreads();
        float x = (h - stats[0]) * stats[1] * eg[tid] + ebe[tid];
        hact[tid] = gelu_exact(x);
        __syncthreads();
    }
    // enhancer layer 2: H->D
    for (int d = tid; d < D_; d += 256) {
        float o = eb2[d];
        for (int j = 0; j < H_; ++j) o += hact[j] * ew2[(size_t)j * D_ + d];
        o *= valid;
        ebuf[d] = o;
        out_enh[(size_t)e * D_ + d] = o;
    }
    __syncthreads();

    // type layer 1: D->H
    float h2 = tb1[tid];
    for (int d = 0; d < D_; ++d) h2 += ebuf[d] * tw1[(size_t)d * H_ + tid];
    {
        float s = h2, q = h2 * h2;
        for (int m = 1; m < 32; m <<= 1) { s += __shfl_xor(s, m, 32); q += __shfl_xor(q, m, 32); }
        if ((tid & 31) == 0) { red[tid >> 5] = s; red[8 + (tid >> 5)] = q; }
        __syncthreads();
        if (tid == 0) {
            float ss = 0.f, qq = 0.f;
            for (int i = 0; i < 8; ++i) { ss += red[i]; qq += red[8 + i]; }
            float mu = ss / H_;
            stats[0] = mu; stats[1] = rsqrtf(qq / H_ - mu * mu + EPS_);
        }
        __syncthreads();
        float x = (h2 - stats[0]) * stats[1] * tg[tid] + tbe[tid];
        hact[tid] = gelu_exact(x);
        __syncthreads();
    }
    // type layer 2: H->T, softmax, argmax
    if (tid < T_) {
        float o = tb2[tid];
        for (int j = 0; j < H_; ++j) o += hact[j] * tw2[(size_t)j * T_ + tid];
        lg[tid] = o;
        out_logits[(size_t)e * T_ + tid] = o;
    }
    __syncthreads();
    if (tid == 0) {
        float mx = lg[0]; int am = 0;
        for (int t = 1; t < T_; ++t) if (lg[t] > mx) { mx = lg[t]; am = t; }
        float ex[T_], se = 0.f;
        for (int t = 0; t < T_; ++t) { ex[t] = expf(lg[t] - mx); se += ex[t]; }
        float inv = 1.f / se;
        for (int t = 0; t < T_; ++t) out_probs[(size_t)e * T_ + t] = ex[t] * inv;
        out_types[e] = (float)am;
    }
}

// ---------------------------------------------------------------------------
extern "C" void kernel_launch(void* const* d_in, const int* in_sizes, int n_in,
                              void* d_out, int out_size, void* d_ws, size_t ws_size,
                              hipStream_t stream) {
    const float* X    = (const float*)d_in[0];
    const int*   msk  = (const int*)d_in[1];
    const float* sw1  = (const float*)d_in[2];
    const float* sb1  = (const float*)d_in[3];
    const float* sg   = (const float*)d_in[4];
    const float* sbe  = (const float*)d_in[5];
    const float* sw2  = (const float*)d_in[6];
    const float* sb2  = (const float*)d_in[7];
    const float* ew1  = (const float*)d_in[8];
    const float* eb1  = (const float*)d_in[9];
    const float* eg   = (const float*)d_in[10];
    const float* ebe  = (const float*)d_in[11];
    const float* ew2  = (const float*)d_in[12];
    const float* eb2  = (const float*)d_in[13];
    const float* tw1  = (const float*)d_in[14];
    const float* tb1  = (const float*)d_in[15];
    const float* tg   = (const float*)d_in[16];
    const float* tbe  = (const float*)d_in[17];
    const float* tw2  = (const float*)d_in[18];
    const float* tb2  = (const float*)d_in[19];

    const int BK = B_ * K_ENT;
    float* o = (float*)d_out;
    float* o_scores = o;                       // B*S
    float* o_enh    = o_scores + B_ * S_;      // B*K*D
    float* o_logits = o_enh + (size_t)BK * D_; // B*K*T
    float* o_escore = o_logits + BK * T_;      // B*K
    float* o_probs  = o_escore + BK;           // B*K*T
    float* o_types  = o_probs + BK * T_;       // B*K
    float* o_validk = o_types + BK;            // B*K

    char* ws = (char*)d_ws;
    int*   metaStart = (int*)ws;               ws += BK * sizeof(int);
    int*   metaLen   = (int*)ws;               ws += BK * sizeof(int);
    float* metaValid = (float*)ws;             ws += BK * sizeof(float);
    float* reps      = (float*)ws;             ws += (size_t)BK * D_ * sizeof(float);
    __bf16* w1t      = (__bf16*)ws;            // D*H bf16, 16B-aligned

    cvt_w1_kernel<<<(D_ * H_) / 256, 256, 0, stream>>>(sw1, w1t);
    scorer_kernel<<<(B_ * S_) / 64, 256, 0, stream>>>(
        X, msk, w1t, sb1, sg, sbe, sw2, sb2, o_scores);
    span_kernel<<<B_, 256, 0, stream>>>(
        o_scores, msk, o_escore, o_validk, metaStart, metaLen, metaValid);
    pool_kernel<<<BK, 256, 0, stream>>>(X, metaStart, metaLen, metaValid, reps);
    mlp_kernel<<<BK, 256, 0, stream>>>(
        reps, metaValid, ew1, eb1, eg, ebe, ew2, eb2,
        tw1, tb1, tg, tbe, tw2, tb2,
        o_enh, o_logits, o_probs, o_types);
}
